// OptiMetal3B_27307402068531
// MI455X (gfx1250) — matrix operand order repaired and checked by reading, not compiled, not run on hardware
//
#include <hip/hip_runtime.h>
#include <hip/hip_bf16.h>
#include <cstdint>

// ---------------------------------------------------------------------------
// CDNA5 / gfx1250 implementation of the metal-spectra GNN reference.
// Dense layers -> v_wmma_f32_16x16x32_f16 (f16 in, f32 accumulate).
// A-tile staging -> Tensor Data Mover (tensor_load_to_lds) with HW padding.
// LDS tiles laid out so every WMMA fragment is exactly two ds_load_b128.
// Segment softmax / scatter -> f32 atomic kernels (latency bound anyway).
// ---------------------------------------------------------------------------

typedef __attribute__((ext_vector_type(16))) _Float16 v16h;
typedef __attribute__((ext_vector_type(8)))  _Float16 v8h;
typedef __attribute__((ext_vector_type(4)))  _Float16 v4h;
typedef __attribute__((ext_vector_type(8)))  float    v8f;
typedef _Float16 h16;

#define HID   256
#define HEADS 4
#define DHD   64

#if __has_builtin(__builtin_amdgcn_tensor_load_to_lds)
#define HAVE_TDM 1
#else
#define HAVE_TDM 0
#endif

// ---------------- small helpers ----------------

__device__ __forceinline__ void atomicMaxF(float* addr, float val) {
    unsigned int* ua = (unsigned int*)addr;
    unsigned int cur = __float_as_uint(*addr);
    while (__uint_as_float(cur) < val) {
        unsigned int old = atomicCAS(ua, cur, __float_as_uint(val));
        if (old == cur) break;
        cur = old;
    }
}

__global__ void k_fill(float* __restrict__ p, float v, size_t n) {
    size_t i = (size_t)blockIdx.x * blockDim.x + threadIdx.x;
    if (i < n) p[i] = v;
}

__global__ void k_cvt(const float* __restrict__ s, h16* __restrict__ d, size_t n) {
    size_t i = (size_t)blockIdx.x * blockDim.x + threadIdx.x;
    if (i < n) d[i] = (h16)s[i];
}

// ---------------- geometry / feature kernels ----------------

__global__ void k_edge_geom(const float* __restrict__ pos, const float* __restrict__ lat,
                            const int* __restrict__ ei, const float* __restrict__ pbc,
                            const int* __restrict__ eb,
                            float* __restrict__ evec, float* __restrict__ elen,
                            h16* __restrict__ rbf, int E) {
    int m = blockIdx.x * blockDim.x + threadIdx.x;
    if (m >= E) return;
    int s = ei[m], d = ei[E + m];
    const float* L = lat + (size_t)eb[m] * 9;
    float ox = pbc[3*m+0], oy = pbc[3*m+1], oz = pbc[3*m+2];
    float vx = ox*L[0] + oy*L[3] + oz*L[6] + pos[3*d+0] - pos[3*s+0];
    float vy = ox*L[1] + oy*L[4] + oz*L[7] + pos[3*d+1] - pos[3*s+1];
    float vz = ox*L[2] + oy*L[5] + oz*L[8] + pos[3*d+2] - pos[3*s+2];
    evec[3*m+0] = vx; evec[3*m+1] = vy; evec[3*m+2] = vz;
    float r = sqrtf(vx*vx + vy*vy + vz*vz);
    elen[m] = r;
    #pragma unroll 8
    for (int i = 0; i < 64; ++i) {
        float c = (5.5f / 63.f) * (float)i;        // linspace(0, R_MAX, 64)
        float t = (r - c) * 0.5f;                  // / RBF_WIDTH (2.0)
        rbf[(size_t)m*64 + i] = (h16)__expf(-t*t);
    }
}

__global__ void k_angle(const int* __restrict__ tb, const float* __restrict__ evec,
                        const float* __restrict__ elen, h16* __restrict__ ang, int A) {
    int m = blockIdx.x * blockDim.x + threadIdx.x;
    if (m >= A) return;
    int j = tb[m], k = tb[A + m];
    float dot = evec[3*j+0]*evec[3*k+0] + evec[3*j+1]*evec[3*k+1] + evec[3*j+2]*evec[3*k+2];
    float c = dot / (elen[j] * elen[k]);
    c = fminf(fmaxf(c, -1.f + 1e-7f), 1.f - 1e-7f);
    float c2 = c * c;
    ang[(size_t)m*4 + 0] = (h16)1.f;
    ang[(size_t)m*4 + 1] = (h16)c;
    ang[(size_t)m*4 + 2] = (h16)(0.5f * (3.f*c2 - 1.f));
    ang[(size_t)m*4 + 3] = (h16)(0.5f * (5.f*c2*c - 3.f*c));
}

__global__ void k_gather_emb(const int* __restrict__ z, const float* __restrict__ emb,
                             h16* __restrict__ out, int N) {
    int i = blockIdx.x * blockDim.x + threadIdx.x;
    if (i >= N * 64) return;
    int row = i >> 6, col = i & 63;
    int zz = z[row]; if (zz < 0) zz = 0; if (zz > 99) zz = 99;
    out[i] = (h16)emb[(size_t)zz*64 + col];
}

// ---------------- TDM descriptor (CDNA5 ISA 08_async_tensor 8.3/8.4) ----------
// Loads a TM x TK 2-byte-element tile at gptr (row stride strideK elements)
// into LDS at lds_off with hardware padding: after every 16 DWORDs (32 halves)
// insert 4 DWORDs (8 halves) -> LDS row stride 40 halves == LDA.
// tensor_dim = remaining (rem_k, rem_m): OOB reads return zero (free edge pad).

__device__ __forceinline__ void tdm_load_tile_a(const h16* gptr, unsigned lds_off,
                                                int rem_k, int rem_m, int strideK) {
#if HAVE_TDM
    typedef unsigned int u32x4 __attribute__((ext_vector_type(4)));
    typedef int i32x8 __attribute__((ext_vector_type(8)));
    typedef int i32x4 __attribute__((ext_vector_type(4)));
    unsigned long long ga = (unsigned long long)gptr;
    u32x4 g0;
    g0[0] = 1u;                                                  // count=1, user D#
    g0[1] = lds_off;                                             // lds_addr (bytes)
    g0[2] = (unsigned)(ga & 0xffffffffu);                        // global_addr[31:0]
    g0[3] = (unsigned)((ga >> 32) & 0x1ffffffu) | (2u << 30);    // addr[56:32], type=2
    unsigned d0 = (unsigned)rem_k;   // tensor_dim0 (elements)
    unsigned d1 = (unsigned)rem_m;   // tensor_dim1 (rows)
    i32x8 g1;
    g1[0] = (int)((1u << 16)        // data_size = 2 bytes
                | (1u << 20)        // pad_enable
                | (3u << 22)        // pad_interval: 16 DWORDs
                | (3u << 25));      // pad_amount:   4 DWORDs
    g1[1] = (int)((d0 & 0xffffu) << 16);                         // dim0[15:0] @63:48
    g1[2] = (int)((d0 >> 16) | ((d1 & 0xffffu) << 16));          // dim0 hi | dim1 lo
    g1[3] = (int)((d1 >> 16) | (32u << 16));                     // dim1 hi | tile_dim0=32
    g1[4] = (int)128u;                                           // tile_dim1=128, tile_dim2=0
    g1[5] = (int)(unsigned)strideK;                              // dim0_stride[31:0]
    g1[6] = 0;                                                   // stride hi, dim1_stride
    g1[7] = 0;
    i32x4 z4 = {0, 0, 0, 0};
    i32x8 z8 = {0, 0, 0, 0, 0, 0, 0, 0};
    // amdgpu-toolchain (clang-23) 6-arg form: (g0, g1, g2, g3, g4, cpol)
    __builtin_amdgcn_tensor_load_to_lds(g0, g1, z4, z4, z8, 0);
#else
    (void)gptr; (void)lds_off; (void)rem_k; (void)rem_m; (void)strideK;
#endif
}

// ---------------- WMMA GEMM ----------------
// C[M,N] = act( A[M,K] @ B[K,N] + bias ), A/B f16 row-major, acc f32.
// Block = 256 threads = 8 wave32 waves. Tile 128x64, K-step 32.
// A tile: TDM DMA -> As (row-major, stride LDA=40 halves; 20 bank-words,
//         conflict-free across 16 lanes, 16B aligned for ds_load_b128).
// B tile: VALU-staged TRANSPOSED -> Bt[col][k], stride LDB=40.
// Fragments (ISA 7.12.2):
//   A lane: row = 16w + (lane&15); halves = K[hi*8..+8) ++ K[16+hi*8..+8)
//   B lane: col = 16t + (lane&15); halves = K[hi*8..+8) ++ K[16+hi*8..+8)
//   -> each fragment = two contiguous ds_load_b128.

#define TM  128
#define TN  64
#define TK  32
#define LDA 40
#define LDB 40

__global__ __launch_bounds__(256) void k_gemm(
    const h16* __restrict__ A, const h16* __restrict__ B,
    const float* __restrict__ bias,
    float* __restrict__ Cf, h16* __restrict__ Ch,
    int M, int N, int K, int relu)
{
    __shared__ h16 As[TM * LDA];   // 10240 B
    __shared__ h16 Bt[TN * LDB];   //  5120 B

    int tid  = threadIdx.x;
    int wave = tid >> 5;
    int lane = tid & 31;
    int l    = lane & 15;
    int hi   = lane >> 4;
    int m_blk = blockIdx.x * TM;
    int n_blk = blockIdx.y * TN;

    const bool nvec = ((N & 7) == 0);   // 16B-aligned vector staging of B rows
    const unsigned as_off = (unsigned)(size_t)(const void*)As;  // LDS byte offset

    v8f acc[4] = {};

    for (int kb = 0; kb < K; kb += TK) {
#if HAVE_TDM
        // ---- stage A tile via Tensor Data Mover (wave 0 issues the DMA) ----
        if (wave == 0) {
            tdm_load_tile_a(A + (size_t)m_blk * K + kb, as_off,
                            K - kb, M - m_blk, K);
        }
#else
        // ---- fallback: VALU staging of A tile ----
        {
            int row = tid >> 1;
            int cb  = (tid & 1) * 16;
            int gm  = m_blk + row;
            if (gm < M && kb + cb + 16 <= K) {
                const v8h* src = (const v8h*)(A + (size_t)gm * K + kb + cb);
                *(v8h*)&As[row * LDA + cb]     = src[0];
                *(v8h*)&As[row * LDA + cb + 8] = src[1];
            } else {
                #pragma unroll
                for (int i = 0; i < 16; ++i) {
                    int gk = kb + cb + i;
                    As[row * LDA + cb + i] =
                        (gm < M && gk < K) ? A[(size_t)gm * K + gk] : (h16)0.f;
                }
            }
        }
#endif
        // ---- stage B tile transposed: thread t -> k row t/8, 8 cols at (t&7)*8 ----
        {
            int k  = tid >> 3;
            int cb = (tid & 7) * 8;
            int gk = kb + k;
            if (nvec && gk < K && n_blk + cb + 8 <= N) {
                v8h v = *(const v8h*)(B + (size_t)gk * N + n_blk + cb);
                #pragma unroll
                for (int i = 0; i < 8; ++i) Bt[(cb + i) * LDB + k] = v[i];
            } else {
                #pragma unroll
                for (int i = 0; i < 8; ++i) {
                    int gn = n_blk + cb + i;
                    Bt[(cb + i) * LDB + k] =
                        (gk < K && gn < N) ? B[(size_t)gk * N + gn] : (h16)0.f;
                }
            }
        }
        // hint next-K B tile toward the caches (global_prefetch_b8)
        if (kb + TK < K) {
            int pk = kb + TK + (tid >> 3); if (pk >= K) pk = K - 1;
            __builtin_prefetch(&B[(size_t)pk * N + n_blk], 0, 0);
        }
#if HAVE_TDM
        if (wave == 0) __builtin_amdgcn_s_wait_tensorcnt(0);
#endif
        __syncthreads();

        // ---- A fragment: two contiguous b128 DS loads ----
        union { v16h v; v8h h[2]; } ua;
        {
            int row = wave * 16 + l;
            ua.h[0] = *(const v8h*)&As[row * LDA + hi * 8];
            ua.h[1] = *(const v8h*)&As[row * LDA + 16 + hi * 8];
        }
        // ---- 4 B fragments + WMMA ----
        #pragma unroll
        for (int t = 0; t < 4; ++t) {
            union { v16h v; v8h h[2]; } ub;
            int col = t * 16 + l;
            ub.h[0] = *(const v8h*)&Bt[col * LDB + hi * 8];
            ub.h[1] = *(const v8h*)&Bt[col * LDB + 16 + hi * 8];
            acc[t] = __builtin_amdgcn_wmma_f32_16x16x32_f16(
                false, ua.v, false, ub.v, (short)0, acc[t], false, false);
        }
        __syncthreads();
    }

    // ---- store ----
    #pragma unroll
    for (int t = 0; t < 4; ++t) {
        int col = n_blk + t * 16 + l;
        if (col >= N) continue;
        float bv = bias ? bias[col] : 0.f;
        #pragma unroll
        for (int i = 0; i < 8; ++i) {
            int row = m_blk + wave * 16 + (hi ? 8 : 0) + i;
            if (row < M) {
                float v = acc[t][i] + bv;
                if (relu) v = fmaxf(v, 0.f);
                if (Cf) Cf[(size_t)row * N + col] = v;
                if (Ch) Ch[(size_t)row * N + col] = (h16)v;
            }
        }
    }
}

// ---------------- attention (segment softmax) kernels ----------------

__global__ void k_logits(const h16* __restrict__ Q, const h16* __restrict__ K,
                         const h16* __restrict__ Ek,
                         const int* __restrict__ src, const int* __restrict__ dst,
                         float* __restrict__ logit, float* __restrict__ segmax, int Mmsg) {
    int idx = blockIdx.x * blockDim.x + threadIdx.x;
    if (idx >= Mmsg * HEADS) return;
    int m = idx >> 2, h = idx & 3;
    int s = src[m], d = dst[m];
    const v8h* q = (const v8h*)(Q  + (size_t)d * HID + h * DHD);
    const v8h* k = (const v8h*)(K  + (size_t)s * HID + h * DHD);
    const v8h* e = (const v8h*)(Ek + (size_t)m * HID + h * DHD);
    float acc = 0.f;
    #pragma unroll
    for (int c = 0; c < 8; ++c) {
        v8h qa = q[c], ka = k[c], ea = e[c];
        #pragma unroll
        for (int i = 0; i < 8; ++i)
            acc += (float)qa[i] * ((float)ka[i] + (float)ea[i]);
    }
    acc *= 0.125f;  // 1/sqrt(64)
    logit[idx] = acc;
    atomicMaxF(&segmax[(size_t)d * HEADS + h], acc);
}

__global__ void k_attnexp(float* __restrict__ logit, const float* __restrict__ segmax,
                          float* __restrict__ segsum, const int* __restrict__ dst, int Mmsg) {
    int idx = blockIdx.x * blockDim.x + threadIdx.x;
    if (idx >= Mmsg * HEADS) return;
    int m = idx >> 2, h = idx & 3;
    int d = dst[m];
    float e = __expf(logit[idx] - segmax[(size_t)d * HEADS + h]);
    logit[idx] = e;
    atomicAdd(&segsum[(size_t)d * HEADS + h], e);
}

__global__ void k_scatter(const h16* __restrict__ V, const h16* __restrict__ Ek,
                          const int* __restrict__ src, const int* __restrict__ dst,
                          const float* __restrict__ eexp, const float* __restrict__ segsum,
                          float* __restrict__ agg, int Mmsg) {
    int idx = blockIdx.x * blockDim.x + threadIdx.x;
    if (idx >= Mmsg * 64) return;
    int m  = idx >> 6;
    int c0 = (idx & 63) * 4;
    int h  = c0 >> 6;
    int s = src[m], d = dst[m];
    float alpha = eexp[(size_t)m * HEADS + h] / (segsum[(size_t)d * HEADS + h] + 1e-9f);
    v4h vv = *(const v4h*)(V  + (size_t)s * HID + c0);
    v4h ev = *(const v4h*)(Ek + (size_t)m * HID + c0);
    #pragma unroll
    for (int j = 0; j < 4; ++j) {
        float v = (float)vv[j] + (float)ev[j];
        atomicAdd(&agg[(size_t)d * HID + c0 + j], alpha * v);
    }
}

__global__ void k_finalize(h16* __restrict__ x, const float* __restrict__ agg,
                           const float* __restrict__ skip, size_t n) {
    size_t i = (size_t)blockIdx.x * blockDim.x + threadIdx.x;
    if (i >= n) return;
    x[i] = (h16)((float)x[i] + fmaxf(agg[i] + skip[i], 0.f));
}

// ---------------- pooling kernels ----------------

__global__ void k_poolmax(const float* __restrict__ gate, const int* __restrict__ nb,
                          float* __restrict__ gmax, int N) {
    int i = blockIdx.x * blockDim.x + threadIdx.x;
    if (i >= N * HID) return;
    int g = nb[i >> 8];
    atomicMaxF(&gmax[(size_t)g * HID + (i & 255)], gate[i]);
}

__global__ void k_poolexp(float* __restrict__ gate, const int* __restrict__ nb,
                          const float* __restrict__ gmax, float* __restrict__ gsum, int N) {
    int i = blockIdx.x * blockDim.x + threadIdx.x;
    if (i >= N * HID) return;
    int g = nb[i >> 8];
    float e = __expf(gate[i] - gmax[(size_t)g * HID + (i & 255)]);
    gate[i] = e;
    atomicAdd(&gsum[(size_t)g * HID + (i & 255)], e);
}

__global__ void k_poolsum(const float* __restrict__ gate, const h16* __restrict__ x,
                          const int* __restrict__ nb, const float* __restrict__ gsum,
                          float* __restrict__ pooled, int N) {
    int i = blockIdx.x * blockDim.x + threadIdx.x;
    if (i >= N * HID) return;
    int g = nb[i >> 8];
    float alpha = gate[i] / (gsum[(size_t)g * HID + (i & 255)] + 1e-9f);
    atomicAdd(&pooled[(size_t)g * HID + (i & 255)], alpha * (float)x[i]);
}

// ---------------- host helpers ----------------

static inline void fillf(hipStream_t st, float* p, float v, size_t n) {
    k_fill<<<(unsigned)((n + 255) / 256), 256, 0, st>>>(p, v, n);
}
static inline void cvt16(hipStream_t st, const float* s, h16* d, size_t n) {
    k_cvt<<<(unsigned)((n + 255) / 256), 256, 0, st>>>(s, d, n);
}
static inline void gemm(hipStream_t st, const h16* A, const h16* B, const float* bias,
                        float* Cf, h16* Ch, int M, int N, int K, int relu) {
    dim3 g((M + TM - 1) / TM, (N + TN - 1) / TN);
    k_gemm<<<g, 256, 0, st>>>(A, B, bias, Cf, Ch, M, N, K, relu);
}

static void transformer_mp(hipStream_t st,
                           h16* x, int nrows,
                           const int* src, const int* dst, int Mmsg,
                           const h16* eattr,
                           const h16* Wq, const h16* Wk, const h16* Wv,
                           const h16* We, const h16* Ws,
                           h16* Q, h16* Kb, h16* V, h16* Ek,
                           float* skip, float* agg,
                           float* logit, float* segmax, float* segsum) {
    gemm(st, x, Wq, nullptr, nullptr, Q,  nrows, HID, HID, 0);
    gemm(st, x, Wk, nullptr, nullptr, Kb, nrows, HID, HID, 0);
    gemm(st, x, Wv, nullptr, nullptr, V,  nrows, HID, HID, 0);
    gemm(st, eattr, We, nullptr, nullptr, Ek, Mmsg, HID, HID, 0);
    gemm(st, x, Ws, nullptr, skip, nullptr, nrows, HID, HID, 0);
    fillf(st, segmax, -1e30f, (size_t)nrows * HEADS);
    fillf(st, segsum, 0.f,    (size_t)nrows * HEADS);
    fillf(st, agg,    0.f,    (size_t)nrows * HID);
    int t4 = Mmsg * HEADS;
    k_logits <<<(t4 + 255) / 256, 256, 0, st>>>(Q, Kb, Ek, src, dst, logit, segmax, Mmsg);
    k_attnexp<<<(t4 + 255) / 256, 256, 0, st>>>(logit, segmax, segsum, dst, Mmsg);
    int t64 = Mmsg * 64;
    k_scatter<<<(t64 + 255) / 256, 256, 0, st>>>(V, Ek, src, dst, logit, segsum, agg, Mmsg);
    size_t nf = (size_t)nrows * HID;
    k_finalize<<<(unsigned)((nf + 255) / 256), 256, 0, st>>>(x, agg, skip, nf);
}

// ---------------- entry point ----------------

extern "C" void kernel_launch(void* const* d_in, const int* in_sizes, int n_in,
                              void* d_out, int out_size, void* d_ws, size_t ws_size,
                              hipStream_t stream) {
    (void)n_in; (void)out_size; (void)ws_size;

    const int*   atomic_number = (const int*)  d_in[0];
    const float* position      = (const float*)d_in[1];
    const float* lattice       = (const float*)d_in[2];
    const int*   edge_index    = (const int*)  d_in[3];
    const float* pbc_offset    = (const float*)d_in[4];
    const int*   tb_index      = (const int*)  d_in[5];
    const int*   node_batch    = (const int*)  d_in[6];
    const int*   edge_batch    = (const int*)  d_in[7];
    // d_in[8] angle_batch unused (derivable)
    const float* atom_emb      = (const float*)d_in[9];

    const int N = in_sizes[0];
    const int E = in_sizes[3] / 2;
    const int A = in_sizes[5] / 2;
    const int G = in_sizes[2] / 9;

    // --- weight table (params flattened in insertion order) ---
    const int widx[36] = {10,12,14, 16,18,20, 22,24,26,
                          28,29,30,31,32, 33,34,35,36,37,
                          38,39,40,41,42, 43,44,45,46,47,
                          48, 50,52,54, 56,58,60};
    const size_t HH = (size_t)HID * HID;
    const size_t wele[36] = {64*256, HH, HH, 64*256, HH, HH, 4*256, HH, HH,
                             HH,HH,HH,HH,HH, HH,HH,HH,HH,HH,
                             HH,HH,HH,HH,HH, HH,HH,HH,HH,HH,
                             HH, 256*2048, 2048*2048, (size_t)2048*4002,
                             HH, HH, 256};

    // --- workspace bump allocator (256B aligned) ---
    char* wp = (char*)d_ws;
    auto alloc = [&](size_t bytes) -> void* {
        void* r = (void*)wp;
        wp += (bytes + 255) & ~(size_t)255;
        return r;
    };

    h16* w16[36];
    for (int i = 0; i < 36; ++i) w16[i] = (h16*)alloc(wele[i] * sizeof(h16));

    float* evec   = (float*)alloc((size_t)E * 3 * 4);
    float* elen   = (float*)alloc((size_t)E * 4);
    h16*   rbf16  = (h16*)  alloc((size_t)E * 64 * 2);
    h16*   ang16  = (h16*)  alloc((size_t)A * 4 * 2);
    h16*   nemb16 = (h16*)  alloc((size_t)N * 64 * 2);
    h16*   x16    = (h16*)  alloc((size_t)N * HID * 2);
    h16*   e16    = (h16*)  alloc((size_t)E * HID * 2);
    h16*   a16    = (h16*)  alloc((size_t)A * HID * 2);
    h16*   t0     = (h16*)  alloc((size_t)E * HID * 2);
    h16*   t1     = (h16*)  alloc((size_t)E * HID * 2);
    h16*   Qb     = (h16*)  alloc((size_t)E * HID * 2);
    h16*   Kb     = (h16*)  alloc((size_t)E * HID * 2);
    h16*   Vb     = (h16*)  alloc((size_t)E * HID * 2);
    h16*   Ekb    = (h16*)  alloc((size_t)E * HID * 2);
    float* skipf  = (float*)alloc((size_t)E * HID * 4);
    float* aggf   = (float*)alloc((size_t)E * HID * 4);
    float* logit  = (float*)alloc((size_t)E * HEADS * 4);
    float* segmax = (float*)alloc((size_t)E * HEADS * 4);
    float* segsum = (float*)alloc((size_t)E * HEADS * 4);
    float* gatef  = (float*)alloc((size_t)N * HID * 4);
    float* gmax   = (float*)alloc((size_t)G * HID * 4);
    float* gsum   = (float*)alloc((size_t)G * HID * 4);
    float* pooled = (float*)alloc((size_t)G * HID * 4);
    h16*   pool16 = (h16*)  alloc((size_t)G * HID * 2);
    h16*   sh1    = (h16*)  alloc((size_t)G * 2048 * 2);
    h16*   sh2    = (h16*)  alloc((size_t)G * 2048 * 2);
    h16*   dh1    = (h16*)  alloc((size_t)G * HID * 2);
    h16*   dh2    = (h16*)  alloc((size_t)G * HID * 2);

    // --- convert weights to f16 ---
    for (int i = 0; i < 36; ++i)
        cvt16(stream, (const float*)d_in[widx[i]], w16[i], wele[i]);

    // --- geometry + input features ---
    k_edge_geom<<<(E + 255) / 256, 256, 0, stream>>>(position, lattice, edge_index,
                                                     pbc_offset, edge_batch,
                                                     evec, elen, rbf16, E);
    k_angle<<<(A + 255) / 256, 256, 0, stream>>>(tb_index, evec, elen, ang16, A);
    k_gather_emb<<<(N * 64 + 255) / 256, 256, 0, stream>>>(atomic_number, atom_emb, nemb16, N);

    const float* bN0 = (const float*)d_in[11]; const float* bN1 = (const float*)d_in[13];
    const float* bN2 = (const float*)d_in[15];
    const float* bE0 = (const float*)d_in[17]; const float* bE1 = (const float*)d_in[19];
    const float* bE2 = (const float*)d_in[21];
    const float* bA0 = (const float*)d_in[23]; const float* bA1 = (const float*)d_in[25];
    const float* bA2 = (const float*)d_in[27];
    const float* bg  = (const float*)d_in[49];
    const float* bS0 = (const float*)d_in[51]; const float* bS1 = (const float*)d_in[53];
    const float* bS2 = (const float*)d_in[55];
    const float* bD0 = (const float*)d_in[57]; const float* bD1 = (const float*)d_in[59];
    const float* bD2 = (const float*)d_in[61];

    // --- input MLPs (relu on all but last layer) ---
    gemm(stream, nemb16, w16[0], bN0, nullptr, t0,  N, HID, 64,  1);
    gemm(stream, t0,     w16[1], bN1, nullptr, t1,  N, HID, HID, 1);
    gemm(stream, t1,     w16[2], bN2, nullptr, x16, N, HID, HID, 0);

    gemm(stream, rbf16,  w16[3], bE0, nullptr, t0,  E, HID, 64,  1);
    gemm(stream, t0,     w16[4], bE1, nullptr, t1,  E, HID, HID, 1);
    gemm(stream, t1,     w16[5], bE2, nullptr, e16, E, HID, HID, 0);

    gemm(stream, ang16,  w16[6], bA0, nullptr, t0,  A, HID, 4,   1);
    gemm(stream, t0,     w16[7], bA1, nullptr, t1,  A, HID, HID, 1);
    gemm(stream, t1,     w16[8], bA2, nullptr, a16, A, HID, HID, 0);

    // --- triplet blocks: angle->edge MP (line graph), then edge->node MP ---
    const int* tb0 = tb_index;          // dst (edge ij)
    const int* tb1 = tb_index + A;      // src (edge ik)
    const int* ei0 = edge_index;        // src node
    const int* ei1 = edge_index + E;    // dst node
    for (int l = 0; l < 2; ++l) {
        int be = 9 + l * 10;            // edge MP weight slot base
        int bn = 14 + l * 10;           // node MP weight slot base
        transformer_mp(stream, e16, E, tb1, tb0, A, a16,
                       w16[be+0], w16[be+1], w16[be+2], w16[be+3], w16[be+4],
                       Qb, Kb, Vb, Ekb, skipf, aggf, logit, segmax, segsum);
        transformer_mp(stream, x16, N, ei0, ei1, E, e16,
                       w16[bn+0], w16[bn+1], w16[bn+2], w16[bn+3], w16[bn+4],
                       Qb, Kb, Vb, Ekb, skipf, aggf, logit, segmax, segsum);
    }

    // --- vector attention pooling ---
    gemm(stream, x16, w16[29], bg, gatef, nullptr, N, HID, HID, 0);
    fillf(stream, gmax,   -1e30f, (size_t)G * HID);
    fillf(stream, gsum,   0.f,    (size_t)G * HID);
    fillf(stream, pooled, 0.f,    (size_t)G * HID);
    k_poolmax<<<(N * HID + 255) / 256, 256, 0, stream>>>(gatef, node_batch, gmax, N);
    k_poolexp<<<(N * HID + 255) / 256, 256, 0, stream>>>(gatef, node_batch, gmax, gsum, N);
    k_poolsum<<<(N * HID + 255) / 256, 256, 0, stream>>>(gatef, x16, node_batch, gsum, pooled, N);
    cvt16(stream, pooled, pool16, (size_t)G * HID);

    // --- heads ---
    float* out  = (float*)d_out;              // eps: [G, 4002] flat
    float* outd = out + (size_t)G * 4002;     // drude: [G]
    gemm(stream, pool16, w16[30], bS0, nullptr, sh1, G, 2048, HID,  1);
    gemm(stream, sh1,    w16[31], bS1, nullptr, sh2, G, 2048, 2048, 1);
    gemm(stream, sh2,    w16[32], bS2, out,  nullptr, G, 4002, 2048, 0);

    gemm(stream, pool16, w16[33], bD0, nullptr, dh1, G, HID, HID, 1);
    gemm(stream, dh1,    w16[34], bD1, nullptr, dh2, G, HID, HID, 1);
    gemm(stream, dh2,    w16[35], bD2, outd, nullptr, G, 1,   HID, 1);  // final relu fused
}
